// FactorizedAttention_54992761258765
// MI455X (gfx1250) — matrix operand order repaired and checked by reading, not compiled, not run on hardware
//
#include <hip/hip_runtime.h>
#include <math.h>

#define HIDDEN 384
#define NF 16
#define FD 24
#define LN_EPS 1e-5f
#define RS 392   // padded LDS row stride (elements): 784B = 196 dwords, 196%64=4 -> conflict-free

typedef __attribute__((ext_vector_type(16))) __bf16 v16bf;
typedef __attribute__((ext_vector_type(8)))  __bf16 v8bf;
typedef __attribute__((ext_vector_type(8)))  float  v8f;
typedef __attribute__((ext_vector_type(4)))  float  v4f;
typedef __attribute__((ext_vector_type(4)))  int    v4i;

#ifndef __has_builtin
#define __has_builtin(x) 0
#endif

#if __has_builtin(__builtin_amdgcn_global_load_async_to_lds_b128)
#define USE_ASYNC_BUILTIN 1
#else
#define USE_ASYNC_BUILTIN 0
#endif

#if __has_builtin(__builtin_amdgcn_s_wait_asynccnt)
#define USE_WAITASYNC_BUILTIN 1
#else
#define USE_WAITASYNC_BUILTIN 0
#endif

typedef __attribute__((address_space(1))) v4i gv4i_t;
typedef __attribute__((address_space(3))) v4i lv4i_t;

__device__ __forceinline__ void async_load_b128(const void* g, void* l) {
#if USE_ASYNC_BUILTIN
  __builtin_amdgcn_global_load_async_to_lds_b128(
      (gv4i_t*)(unsigned long long)g,
      (lv4i_t*)(unsigned int)(unsigned long long)l, 0, 0);
#else
  unsigned int loff = (unsigned int)(unsigned long long)l;
  asm volatile("global_load_async_to_lds_b128 %0, %1, off"
               :: "v"(loff), "v"(g) : "memory");
#endif
}

template <int N>
__device__ __forceinline__ void wait_async() {
#if USE_WAITASYNC_BUILTIN
  __builtin_amdgcn_s_wait_asynccnt(N);
#else
  asm volatile("s_wait_asynccnt %0" :: "i"(N) : "memory");
#endif
}

__device__ __host__ __forceinline__ unsigned short f2bf_bits(float f) {
  union { float f; unsigned u; } c; c.f = f;
  unsigned u = c.u;
  unsigned r = (u + 0x7FFFu + ((u >> 16) & 1u)) >> 16;
  return (unsigned short)r;
}

// ---------------------------------------------------------------------------
// Prep 1: IW = row-softmax(W_inter)  (16x16)
// ---------------------------------------------------------------------------
__global__ void __launch_bounds__(256) iw_kernel(const float* __restrict__ Wi,
                                                 float* __restrict__ IW) {
  int t = threadIdx.x;
  if (t < 256) {
    int i = t >> 4, j = t & 15;
    float m = -1e30f;
    for (int jj = 0; jj < 16; ++jj) m = fmaxf(m, Wi[i * 16 + jj]);
    float s = 0.f;
    for (int jj = 0; jj < 16; ++jj) s += expf(Wi[i * 16 + jj] - m);
    IW[t] = expf(Wi[i * 16 + j] - m) / s;
  }
}

// ---------------------------------------------------------------------------
// Prep 2: Wb1 = bf16(W_embed)  [n][k] row-major
//         Wb2[n][j*24+d] = bf16( sum_i IW[i][j] * W_agg[n][i*24+d] )
// ---------------------------------------------------------------------------
__global__ void __launch_bounds__(256) wprep_kernel(const float* __restrict__ We,
                                                    const float* __restrict__ Wa,
                                                    const float* __restrict__ IW,
                                                    unsigned short* __restrict__ Wb1,
                                                    unsigned short* __restrict__ Wb2) {
  int idx = blockIdx.x * 256 + threadIdx.x;  // 0 .. 384*384-1
  int n = idx / HIDDEN;
  int k = idx - n * HIDDEN;
  Wb1[idx] = f2bf_bits(We[idx]);
  int j = k / FD, d = k - j * FD;
  float s = 0.f;
#pragma unroll
  for (int i = 0; i < NF; ++i) s += IW[i * 16 + j] * Wa[n * HIDDEN + i * FD + d];
  Wb2[idx] = f2bf_bits(s);
}

// ---------------------------------------------------------------------------
// Fragment helpers (16-bit WMMA layouts per CDNA5 ISA 7.12.2)
// ---------------------------------------------------------------------------
// A 16x32 bf16: lanes 0-15 -> row=lane, K = {0..7, 16..23}; lanes 16-31 -> K={8..15, 24..31}
__device__ __forceinline__ v16bf load_afrag_x(const float* xrow, int lane, int Kb) {
  int koff = (lane < 16) ? 0 : 8;
  const float* p0 = xrow + Kb + koff;
  const float* p1 = xrow + Kb + 16 + koff;
  v4f a0 = *(const v4f*)p0;
  v4f a1 = *(const v4f*)(p0 + 4);
  v4f b0 = *(const v4f*)p1;
  v4f b1 = *(const v4f*)(p1 + 4);
  v16bf r;
#pragma unroll
  for (int i = 0; i < 4; ++i) {
    r[i]      = (__bf16)a0[i];
    r[4 + i]  = (__bf16)a1[i];
    r[8 + i]  = (__bf16)b0[i];
    r[12 + i] = (__bf16)b1[i];
  }
  return r;
}

__device__ __forceinline__ v16bf load_afrag_lds(const __bf16* facRow, int lane, int Kb) {
  int koff = (lane < 16) ? 0 : 8;
  v8bf lo = *(const v8bf*)(facRow + Kb + koff);
  v8bf hi = *(const v8bf*)(facRow + Kb + 16 + koff);
  v16bf r;
#pragma unroll
  for (int i = 0; i < 8; ++i) { r[i] = lo[i]; r[8 + i] = hi[i]; }
  return r;
}

// B 32x16 bf16: lanes 0-15 -> col=lane, K = {0..7, 16..23}; lanes 16-31 -> K={8..15, 24..31}
// Weight tile stored [n][k] in LDS with padded stride RS -> conflict-free b128 loads.
__device__ __forceinline__ v16bf load_bfrag(const __bf16* sBbuf, int lane, int Kb) {
  int n = lane & 15;
  int kh = (lane >> 4) << 3;  // 0 or 8
  const __bf16* p = sBbuf + n * RS + Kb + kh;
  v8bf lo = *(const v8bf*)p;
  v8bf hi = *(const v8bf*)(p + 16);
  v16bf r;
#pragma unroll
  for (int i = 0; i < 8; ++i) { r[i] = lo[i]; r[8 + i] = hi[i]; }
  return r;
}

// Stage one 16x384 bf16 weight column tile (12 KB contiguous in memory) into
// LDS with +16B padding per 768B row. 128 threads x 6 x b128 async chunks.
__device__ __forceinline__ void stage_tile(const __bf16* W, int nt, char* dst, int tid) {
  const char* src = (const char*)(W + (size_t)nt * (16 * HIDDEN));
#pragma unroll
  for (int it = 0; it < 6; ++it) {
    int off = (tid + it * 128) * 16;   // source byte offset, 16B chunks
    int row = off / 768;               // 768B per source row
    async_load_b128(src + off, dst + off + row * 16);
  }
}

// ---------------------------------------------------------------------------
// Fused main kernel: 128 threads = 4 waves, each wave owns a 16-row tile.
// ---------------------------------------------------------------------------
__global__ void __launch_bounds__(128)
fa_main(const float* __restrict__ x,
        const float* __restrict__ b_embed,
        const float* __restrict__ b_agg,
        const float* __restrict__ gamma,
        const float* __restrict__ beta,
        const unsigned short* __restrict__ Wb1u,
        const unsigned short* __restrict__ Wb2u,
        float* __restrict__ out,
        float* __restrict__ scores) {
  extern __shared__ char smem[];
  // bytes: [0,25088) B double-buffer | [25088,75264) factors bf16 (padded)
  //        [75264,125440) h bf16 (padded) | [125440,129536) norm^2 f32
  __bf16* sB   = (__bf16*)smem;
  __bf16* sFac = (__bf16*)(smem + 25088);
  __bf16* sH   = (__bf16*)(smem + 75264);
  float*  sNsq = (float*)(smem + 125440);

  const __bf16* Wb1 = (const __bf16*)Wb1u;
  const __bf16* Wb2 = (const __bf16*)Wb2u;

  const int tid  = threadIdx.x;
  const int wave = tid >> 5;
  const int lane = tid & 31;
  const int rowbase = blockIdx.x * 64 + wave * 16;

  __bf16* myFac = sFac + wave * (16 * RS);
  __bf16* myH   = sH   + wave * (16 * RS);
  float*  myNsq = sNsq + wave * 256;

  // A fragments for GEMM1 live in registers for all 24 N-tiles.
  const float* xrow = x + (size_t)(rowbase + (lane & 15)) * HIDDEN;
  v16bf afrag[12];
#pragma unroll
  for (int kt = 0; kt < 12; ++kt) afrag[kt] = load_afrag_x(xrow, lane, kt * 32);

  const int ncol_lane = lane & 15;
  const int rhalf = (lane >> 4) * 8;

  // ---------------- GEMM1: factors = x @ W_embed^T + b_embed ----------------
  stage_tile(Wb1, 0, (char*)sB, tid);
  for (int nt = 0; nt < 24; ++nt) {
    __bf16* bufC = sB + (nt & 1) * (16 * RS);
    __bf16* bufN = sB + ((nt & 1) ^ 1) * (16 * RS);
    if (nt + 1 < 24) { stage_tile(Wb1, nt + 1, (char*)bufN, tid); wait_async<6>(); }
    else             { wait_async<0>(); }
    __syncthreads();

    // Software-pipelined: B fragment for kt+1 is in flight during wmma(kt).
    v8f acc = {};
    v16bf bcur = load_bfrag(bufC, lane, 0);
#pragma unroll
    for (int kt = 0; kt < 11; ++kt) {
      v16bf bnext = load_bfrag(bufC, lane, (kt + 1) * 32);
      acc = __builtin_amdgcn_wmma_f32_16x16x32_bf16(false, afrag[kt], false, bcur,
                                                    (short)0, acc, false, false);
      bcur = bnext;
    }
    acc = __builtin_amdgcn_wmma_f32_16x16x32_bf16(false, afrag[11], false, bcur,
                                                  (short)0, acc, false, false);

    int ncol = nt * 16 + ncol_lane;
    float bias = b_embed[ncol];
#pragma unroll
    for (int g = 0; g < 8; ++g) {
      int r = g + rhalf;
      myFac[r * RS + ncol] = (__bf16)(acc[g] + bias);
    }
    __syncthreads();
  }

  // ---------------- factor norms (atomic-free, conflict-free) ---------------
  {
    int r = lane & 15;
    int jb = (lane >> 4) * 8;
#pragma unroll
    for (int p = 0; p < 8; ++p) {
      int j = jb + p;
      const __bf16* fp = myFac + r * RS + j * FD;
      v8bf f0 = *(const v8bf*)fp;
      v8bf f1 = *(const v8bf*)(fp + 8);
      v8bf f2 = *(const v8bf*)(fp + 16);
      float s = 0.f;
#pragma unroll
      for (int d = 0; d < 8; ++d) {
        float a = (float)f0[d], b = (float)f1[d], c = (float)f2[d];
        s += a * a + b * b + c * c;
      }
      myNsq[r * 16 + j] = s;
    }
  }

  // ---------------- factor scores: softmax over L2 norms --------------------
  if (lane < 16) {
    int r = lane;
    float nv[16];
    float m = -1e30f;
#pragma unroll
    for (int jj = 0; jj < 16; ++jj) { nv[jj] = sqrtf(myNsq[r * 16 + jj]); m = fmaxf(m, nv[jj]); }
    float s = 0.f;
#pragma unroll
    for (int jj = 0; jj < 16; ++jj) { nv[jj] = expf(nv[jj] - m); s += nv[jj]; }
    float inv = 1.f / s;
    float* srow = scores + (size_t)(rowbase + r) * 16;
#pragma unroll
    for (int jj = 0; jj < 16; ++jj) srow[jj] = nv[jj] * inv;
  }
  __syncthreads();

  // ---------------- GEMM2: h = factors @ W2^T + b_agg -----------------------
  const __bf16* myFacRow = myFac + (lane & 15) * RS;
  stage_tile(Wb2, 0, (char*)sB, tid);
  for (int nt = 0; nt < 24; ++nt) {
    __bf16* bufC = sB + (nt & 1) * (16 * RS);
    __bf16* bufN = sB + ((nt & 1) ^ 1) * (16 * RS);
    if (nt + 1 < 24) { stage_tile(Wb2, nt + 1, (char*)bufN, tid); wait_async<6>(); }
    else             { wait_async<0>(); }
    __syncthreads();

    v8f acc = {};
    v16bf acur = load_afrag_lds(myFacRow, lane, 0);
    v16bf bcur = load_bfrag(bufC, lane, 0);
#pragma unroll
    for (int kt = 0; kt < 11; ++kt) {
      v16bf anext = load_afrag_lds(myFacRow, lane, (kt + 1) * 32);
      v16bf bnext = load_bfrag(bufC, lane, (kt + 1) * 32);
      acc = __builtin_amdgcn_wmma_f32_16x16x32_bf16(false, acur, false, bcur,
                                                    (short)0, acc, false, false);
      acur = anext;
      bcur = bnext;
    }
    acc = __builtin_amdgcn_wmma_f32_16x16x32_bf16(false, acur, false, bcur,
                                                  (short)0, acc, false, false);

    int ncol = nt * 16 + ncol_lane;
    float bias = b_agg[ncol];
#pragma unroll
    for (int g = 0; g < 8; ++g) {
      int r = g + rhalf;
      myH[r * RS + ncol] = (__bf16)(acc[g] + bias);
    }
    __syncthreads();
  }

  // -------- LayerNorm + ReLU: in-register stats + wave shuffle reduce -------
  const float invH = 1.f / (float)HIDDEN;
#pragma unroll 1
  for (int r = 0; r < 16; ++r) {
    const __bf16* hrow = myH + r * RS;
    float vals[12];
    float s = 0.f, sq = 0.f;
#pragma unroll
    for (int i = 0; i < 12; ++i) {
      float v = (float)hrow[lane + i * 32];
      vals[i] = v;
      s += v;
      sq += v * v;
    }
#pragma unroll
    for (int off = 16; off > 0; off >>= 1) {
      s  += __shfl_xor(s,  off, 32);
      sq += __shfl_xor(sq, off, 32);
    }
    float mu = s * invH;
    float var = sq * invH - mu * mu;
    float rstd = rsqrtf(var + LN_EPS);
    float* orow = out + (size_t)(rowbase + r) * HIDDEN;
#pragma unroll
    for (int i = 0; i < 12; ++i) {
      int c = lane + i * 32;
      float v = (vals[i] - mu) * rstd * gamma[c] + beta[c];
      orow[c] = fmaxf(v, 0.f);
    }
  }
}

// ---------------------------------------------------------------------------
extern "C" void kernel_launch(void* const* d_in, const int* in_sizes, int n_in,
                              void* d_out, int out_size, void* d_ws, size_t ws_size,
                              hipStream_t stream) {
  const float* x       = (const float*)d_in[0];
  const float* W_embed = (const float*)d_in[1];
  const float* b_embed = (const float*)d_in[2];
  const float* W_inter = (const float*)d_in[3];
  const float* W_agg   = (const float*)d_in[4];
  const float* b_agg   = (const float*)d_in[5];
  const float* gamma   = (const float*)d_in[6];
  const float* beta    = (const float*)d_in[7];
  int B = in_sizes[0] / HIDDEN;

  unsigned short* Wb1 = (unsigned short*)d_ws;                    // 384*384 bf16
  unsigned short* Wb2 = Wb1 + HIDDEN * HIDDEN;                    // 384*384 bf16
  float* IW = (float*)(Wb2 + HIDDEN * HIDDEN);                    // 16*16 f32

  iw_kernel<<<1, 256, 0, stream>>>(W_inter, IW);
  wprep_kernel<<<(HIDDEN * HIDDEN) / 256, 256, 0, stream>>>(W_embed, W_agg, IW, Wb1, Wb2);

  float* out = (float*)d_out;
  float* scores = out + (size_t)B * HIDDEN;
  int blocks = B / 64;            // 4 waves x 16 rows per block
  size_t smem_bytes = 129536;     // see layout in fa_main
  fa_main<<<blocks, 128, smem_bytes, stream>>>(x, b_embed, b_agg, gamma, beta,
                                               (const unsigned short*)Wb1,
                                               (const unsigned short*)Wb2,
                                               out, scores);
}
